// RNN_83391085019707
// MI455X (gfx1250) — matrix operand order repaired and checked by reading, not compiled, not run on hardware
//
#include <hip/hip_runtime.h>

// ---------------- problem constants ----------------
#define T_  512
#define B_  64
#define D_  1024
#define H_  1024
#define L_  2

#define NWG       16          // persistent workgroups (co-resident)
#define NTHREADS  256         // 8 wave32 per WG -> 128 waves; 2 N-tiles per wave
#define WAVES_PER_WG 8

typedef __attribute__((ext_vector_type(16))) __bf16  v16bf;
typedef __attribute__((ext_vector_type(8)))  float   v8f;
typedef __attribute__((ext_vector_type(4)))  unsigned v4u;

// fp32 -> bf16 round-to-nearest-even (bit trick)
__device__ __forceinline__ unsigned short f2bf(float f) {
    unsigned u = __builtin_bit_cast(unsigned, f);
    u += 0x7FFFu + ((u >> 16) & 1u);
    return (unsigned short)(u >> 16);
}

// A fragment: lane<16 -> row m0+lane, K {k..k+7} and {k+16..k+23}
//             lane>=16 -> same row set, K {k+8..k+15} and {k+24..k+31}
// caller pre-offsets base by row*ld + (lane>>4)*8
__device__ __forceinline__ v16bf load_frag_a(const unsigned short* p) {
    union { v4u q[2]; v16bf v; } u;
    u.q[0] = *(const v4u*)(p);
    u.q[1] = *(const v4u*)(p + 16);
    return u.v;
}

// B fragment (B[k,j] = W[j,k], W row-major): lane n holds column j0+(n&15),
// K {k..k+7},{k+16..k+23} for n<16, {k+8..k+15},{k+24..k+31} for n>=16.
// caller pre-offsets base by (j0+(lane&15))*ld + (lane>>4)*8
__device__ __forceinline__ v16bf load_frag_b(const unsigned short* p) {
    union { v4u q[2]; v16bf v; } u;
    u.q[0] = *(const v4u*)(p);
    u.q[1] = *(const v4u*)(p + 16);
    return u.v;
}

__device__ __forceinline__ v8f wmma_bf16(v16bf a, v16bf b, v8f c) {
    return __builtin_amdgcn_wmma_f32_16x16x32_bf16(false, a, false, b, (short)0, c,
                                                   false, false);
}

// Two 16x16 C tiles sharing the A fragment (16x32 macro-tile), K multiple of 32.
// B1 = B0 + 16 rows of W. 3 b128-load-pairs per 2 WMMAs.
__device__ __forceinline__ void gemm_tile2(v8f& c0, v8f& c1,
                                           const unsigned short* __restrict__ Abase,
                                           const unsigned short* __restrict__ B0,
                                           const unsigned short* __restrict__ B1,
                                           int K) {
#pragma unroll 4
    for (int k = 0; k < K; k += 32) {
        __builtin_prefetch(B0 + k + 512, 0, 3);   // WGP-scope prefetch on weight stream
        __builtin_prefetch(B1 + k + 512, 0, 3);
        v16bf a  = load_frag_a(Abase + k);
        v16bf b0 = load_frag_b(B0 + k);
        v16bf b1 = load_frag_b(B1 + k);
        c0 = wmma_bf16(a, b0, c0);
        c1 = wmma_bf16(a, b1, c1);
    }
}

// sense-reversing grid barrier: bar[0]=arrive count, bar[1]=generation
__device__ __forceinline__ void grid_barrier(unsigned* bar) {
    __syncthreads();
    if (threadIdx.x == 0) {
        __threadfence();
        unsigned gen = __hip_atomic_load(&bar[1], __ATOMIC_RELAXED, __HIP_MEMORY_SCOPE_AGENT);
        unsigned arrived = __hip_atomic_fetch_add(&bar[0], 1u, __ATOMIC_ACQ_REL, __HIP_MEMORY_SCOPE_AGENT);
        if (arrived == (unsigned)(NWG - 1)) {
            __hip_atomic_store(&bar[0], 0u, __ATOMIC_RELAXED, __HIP_MEMORY_SCOPE_AGENT);
            __hip_atomic_store(&bar[1], gen + 1u, __ATOMIC_RELEASE, __HIP_MEMORY_SCOPE_AGENT);
        } else {
            while (__hip_atomic_load(&bar[1], __ATOMIC_ACQUIRE, __HIP_MEMORY_SCOPE_AGENT) == gen) {
                __builtin_amdgcn_s_sleep(2);
            }
        }
        __threadfence();
    }
    __syncthreads();
}

// one-time (per call) init: weights fp32->bf16, zero hidden ping buffer, reset barrier
__global__ void rnn_init(const float* __restrict__ Wi, const float* __restrict__ Wh,
                         const float* __restrict__ Wo,
                         unsigned short* __restrict__ wiB, unsigned short* __restrict__ whB,
                         unsigned short* __restrict__ woB,
                         unsigned short* __restrict__ hidA, unsigned* __restrict__ bar) {
    size_t stride = (size_t)gridDim.x * blockDim.x;
    size_t g = (size_t)blockIdx.x * blockDim.x + threadIdx.x;
    const size_t NW = (size_t)L_ * H_ * D_;
    for (size_t i = g; i < NW; i += stride) {
        wiB[i] = f2bf(Wi[i]);
        whB[i] = f2bf(Wh[i]);
        woB[i] = f2bf(Wo[i]);
    }
    for (size_t i = g; i < (size_t)B_ * H_; i += stride) hidA[i] = 0;
    if (g == 0) { bar[0] = 0u; bar[1] = 0u; }
}

__global__ __launch_bounds__(NTHREADS, 1)
void rnn_persistent(const float* __restrict__ x,
                    const float* __restrict__ bi, const float* __restrict__ bh,
                    const float* __restrict__ bo,
                    const unsigned short* __restrict__ wiB,
                    const unsigned short* __restrict__ whB,
                    const unsigned short* __restrict__ woB,
                    unsigned short* __restrict__ actX, unsigned short* __restrict__ actM,
                    unsigned short* __restrict__ hidA, unsigned short* __restrict__ hidB,
                    unsigned short* __restrict__ hbuf,
                    float* __restrict__ out, float* __restrict__ finalh,
                    unsigned* __restrict__ bar) {
    const int tid  = threadIdx.x;
    const int lane = tid & 31;
    const int w    = blockIdx.x * WAVES_PER_WG + (tid >> 5);   // 0..127
    const int m0   = (w >> 5) * 16;                            // 4 M-tiles
    const int j0   = (w & 31) * 32;                            // 32 N-tile pairs
    const int col  = lane & 15;
    const int hi8  = (lane >> 4) * 8;
    const int kA   = hi8;                 // A frag k sub-offset (0 or 8)
    const int kB   = (lane >> 4) * 8;     // B frag k sub-offset (0 or 8)
    const int rowA = m0 + col;            // A frag row for this lane
    const int gtid    = blockIdx.x * NTHREADS + tid;
    const int gstride = NWG * NTHREADS;

    for (int t = 0; t < T_; ++t) {
        // stage x[t] as bf16 layer-0 input
        const float* xt = x + (size_t)t * B_ * D_;
        for (int i = gtid; i < B_ * D_; i += gstride) actX[i] = f2bf(xt[i]);
        grid_barrier(bar);

        for (int l = 0; l < L_; ++l) {
            const int step = t * 2 + l;
            const unsigned short* actIn  = l ? actM : actX;
            const unsigned short* hidIn  = (step & 1) ? hidB : hidA;
            unsigned short*       hidOut = (step & 1) ? hidA : hidB;

            const size_t wOff = (size_t)l * H_ * D_ + (size_t)(j0 + col) * D_ + kB;
            const size_t wStep16 = (size_t)16 * D_;   // 16 weight rows = next N tile

            // ---- phase 1: h = actIn@Wi^T + hidIn@Wh^T + bi + bh ; hidden' = tanh(h)
            v8f c0 = {}, c1 = {};
            gemm_tile2(c0, c1, actIn + (size_t)rowA * D_ + kA,
                       wiB + wOff, wiB + wOff + wStep16, D_);
            gemm_tile2(c0, c1, hidIn + (size_t)rowA * H_ + kA,
                       whB + wOff, whB + wOff + wStep16, H_);
            const float bias0 = bi[l * H_ + j0 + col]      + bh[l * H_ + j0 + col];
            const float bias1 = bi[l * H_ + j0 + 16 + col] + bh[l * H_ + j0 + 16 + col];
            const bool last_step = (step == 2 * T_ - 1);
#pragma unroll
            for (int r = 0; r < 8; ++r) {
                const int row = m0 + hi8 + r;
                const float hv0 = c0[r] + bias0;
                const float hv1 = c1[r] + bias1;
                hbuf[row * H_ + j0 + col]      = f2bf(hv0);
                hbuf[row * H_ + j0 + 16 + col] = f2bf(hv1);
                const float th0 = tanhf(hv0);
                const float th1 = tanhf(hv1);
                hidOut[row * H_ + j0 + col]      = f2bf(th0);
                hidOut[row * H_ + j0 + 16 + col] = f2bf(th1);
                if (last_step) {
                    finalh[row * H_ + j0 + col]      = th0;
                    finalh[row * H_ + j0 + 16 + col] = th1;
                }
            }
            grid_barrier(bar);

            // ---- phase 2: out = h@Wo^T + bo
            v8f o0 = {}, o1 = {};
            gemm_tile2(o0, o1, hbuf + (size_t)rowA * H_ + kA,
                       woB + wOff, woB + wOff + wStep16, H_);
            const float ob0 = bo[l * H_ + j0 + col];
            const float ob1 = bo[l * H_ + j0 + 16 + col];
            if (l == L_ - 1) {
                float* outT = out + (size_t)t * B_ * H_;
#pragma unroll
                for (int r = 0; r < 8; ++r) {
                    const int row = m0 + hi8 + r;
                    outT[row * H_ + j0 + col]      = o0[r] + ob0;
                    outT[row * H_ + j0 + 16 + col] = o1[r] + ob1;
                }
            } else {
#pragma unroll
                for (int r = 0; r < 8; ++r) {
                    const int row = m0 + hi8 + r;
                    actM[row * H_ + j0 + col]      = f2bf(o0[r] + ob0);
                    actM[row * H_ + j0 + 16 + col] = f2bf(o1[r] + ob1);
                }
            }
            grid_barrier(bar);
        }
    }
}

extern "C" void kernel_launch(void* const* d_in, const int* in_sizes, int n_in,
                              void* d_out, int out_size, void* d_ws, size_t ws_size,
                              hipStream_t stream) {
    const float* x  = (const float*)d_in[0];
    const float* Wi = (const float*)d_in[1];
    const float* bi = (const float*)d_in[2];
    const float* Wh = (const float*)d_in[3];
    const float* bh = (const float*)d_in[4];
    const float* Wo = (const float*)d_in[5];
    const float* bo = (const float*)d_in[6];

    float* out    = (float*)d_out;                       // outs [T,B,H]
    float* finalh = out + (size_t)T_ * B_ * H_;          // final_hidden [B,H]

    // workspace layout (bytes)
    char* ws = (char*)d_ws;
    const size_t szW   = (size_t)L_ * H_ * D_ * sizeof(unsigned short); // 4 MB each
    const size_t szAct = (size_t)B_ * H_ * sizeof(unsigned short);      // 128 KB
    unsigned short* wiB  = (unsigned short*)(ws);
    unsigned short* whB  = (unsigned short*)(ws + szW);
    unsigned short* woB  = (unsigned short*)(ws + 2 * szW);
    unsigned short* actX = (unsigned short*)(ws + 3 * szW);
    unsigned short* actM = (unsigned short*)(ws + 3 * szW + szAct);
    unsigned short* hidA = (unsigned short*)(ws + 3 * szW + 2 * szAct);
    unsigned short* hidB = (unsigned short*)(ws + 3 * szW + 3 * szAct);
    unsigned short* hbuf = (unsigned short*)(ws + 3 * szW + 4 * szAct);
    unsigned*       bar  = (unsigned*)      (ws + 3 * szW + 5 * szAct);

    rnn_init<<<2048, 256, 0, stream>>>(Wi, Wh, Wo, wiB, whB, woB, hidA, bar);
    rnn_persistent<<<NWG, NTHREADS, 0, stream>>>(x, bi, bh, bo, wiB, whB, woB,
                                                 actX, actM, hidA, hidB, hbuf,
                                                 out, finalh, bar);
}
